// NonLocalBlock1D_962072674578
// MI455X (gfx1250) — compile-verified, matrix operand order checked by reading, not run on hardware
//
#include <hip/hip_runtime.h>

// ---------------------------------------------------------------------------
// NonLocalBlock1D (B=8, C=512, IC=256, T=2048) for gfx1250 (MI455X, wave32)
// Flash-attention fused pipeline: v_wmma_f32_16x16x32_bf16 +
// global_load_async_to_lds_b128 double-buffered K/V staging.
// ---------------------------------------------------------------------------

typedef __attribute__((ext_vector_type(16))) __bf16 v16bf;
typedef __attribute__((ext_vector_type(8)))  __bf16 v8bf;
typedef __attribute__((ext_vector_type(8)))  float  v8f;

constexpr int Bb  = 8;
constexpr int Cc  = 512;
constexpr int ICc = 256;
constexpr int Tt  = 2048;

#define WMMA_BF16(A, Bm, Cacc) \
  __builtin_amdgcn_wmma_f32_16x16x32_bf16(false, (A), false, (Bm), (short)0, (Cacc), false, false)

// ---- CDNA5 async memory->LDS DMA (tracked by ASYNCcnt) --------------------
__device__ __forceinline__ void async_ld_b128(unsigned lds_addr, const void* gaddr) {
  asm volatile("global_load_async_to_lds_b128 %0, %1, off"
               :: "v"(lds_addr), "v"(gaddr)
               : "memory");
}

__device__ __forceinline__ void wait_async0() {
#if __has_builtin(__builtin_amdgcn_s_wait_asynccnt)
  __builtin_amdgcn_s_wait_asynccnt(0);
#else
  asm volatile("s_wait_asynccnt 0x0" ::: "memory");
#endif
}

__device__ __forceinline__ unsigned lds_off(const void* p) {
  return (unsigned)(size_t)p;  // low 32 bits of flat LDS address == LDS offset
}

// ---------------------------------------------------------------------------
// fp32 -> bf16 elementwise (weights)
// ---------------------------------------------------------------------------
__global__ void f32_to_bf16_kernel(const float* __restrict__ in,
                                   __bf16* __restrict__ out, int n) {
  int i = blockIdx.x * blockDim.x + threadIdx.x;
  if (i < n) out[i] = (__bf16)in[i];
}

// ---------------------------------------------------------------------------
// x (B,C,T) fp32 -> xT (B,T,C) bf16, tiled transpose through LDS
// block (32,8), grid (T/32, C/32, B)
// ---------------------------------------------------------------------------
__global__ __launch_bounds__(256) void transpose_x_kernel(
    const float* __restrict__ x, __bf16* __restrict__ xT) {
  __shared__ float tile[32][33];
  const int b  = blockIdx.z;
  const int c0 = blockIdx.y * 32;
  const int t0 = blockIdx.x * 32;
  const int tx = threadIdx.x, ty = threadIdx.y;
#pragma unroll
  for (int k = 0; k < 4; ++k)
    tile[ty + 8 * k][tx] = x[((size_t)(b * Cc + c0 + ty + 8 * k)) * Tt + t0 + tx];
  __syncthreads();
#pragma unroll
  for (int k = 0; k < 4; ++k)
    xT[((size_t)(b * Tt + t0 + ty + 8 * k)) * Cc + c0 + tx] =
        (__bf16)tile[tx][ty + 8 * k];
}

// ---------------------------------------------------------------------------
// Projections: theta/phi/g = W (IC x C) * x (C x T) + bias, per batch.
// One wave per 16x16 output tile; B fragment (xT) reused across 3 WMMAs.
//   Q  (B,T,IC) <- theta^T   Kb (B,T,IC) <- phi^T   Vg (B,IC,T) <- g
// ---------------------------------------------------------------------------
__global__ __launch_bounds__(256) void proj_kernel(
    const __bf16* __restrict__ xT,
    const __bf16* __restrict__ wg, const float* __restrict__ bg,
    const __bf16* __restrict__ wt, const float* __restrict__ bt,
    const __bf16* __restrict__ wp, const float* __restrict__ bp,
    __bf16* __restrict__ Q, __bf16* __restrict__ Kb, __bf16* __restrict__ Vg) {
  const int lane = threadIdx.x & 31;
  const int wave = threadIdx.x >> 5;
  const int wid  = blockIdx.x * 8 + wave;
  const int b    = wid >> 11;                   // (T/16)*(IC/16) = 2048 / batch
  const int rem  = wid & 2047;
  const int tIdx = rem >> 4;
  const int mIdx = rem & 15;
  const int t0 = tIdx * 16, m0 = mIdx * 16;
  const int lr = lane & 15, h = lane >> 4;

  v8f accT = {}, accP = {}, accG = {};
  const __bf16* xrow  = xT + ((size_t)b * Tt + t0 + lr) * Cc + 16 * h;
  const __bf16* wtrow = wt + (size_t)(m0 + lr) * Cc + 16 * h;
  const __bf16* wprow = wp + (size_t)(m0 + lr) * Cc + 16 * h;
  const __bf16* wgrow = wg + (size_t)(m0 + lr) * Cc + 16 * h;
#pragma unroll
  for (int k0 = 0; k0 < Cc; k0 += 32) {
    v16bf bx = *(const v16bf*)(xrow + k0);
    v16bf at = *(const v16bf*)(wtrow + k0);
    v16bf ap = *(const v16bf*)(wprow + k0);
    v16bf ag = *(const v16bf*)(wgrow + k0);
    accT = WMMA_BF16(at, bx, accT);
    accP = WMMA_BF16(ap, bx, accP);
    accG = WMMA_BF16(ag, bx, accG);
  }
  v8bf q8, k8;
#pragma unroll
  for (int r = 0; r < 8; ++r) {
    const int m = m0 + 8 * h + r;
    q8[r] = (__bf16)(accT[r] + bt[m]);
    k8[r] = (__bf16)(accP[r] + bp[m]);
  }
  *(v8bf*)(Q  + ((size_t)b * Tt + t0 + lr) * ICc + m0 + 8 * h) = q8;
  *(v8bf*)(Kb + ((size_t)b * Tt + t0 + lr) * ICc + m0 + 8 * h) = k8;
#pragma unroll
  for (int r = 0; r < 8; ++r) {
    const int m = m0 + 8 * h + r;
    Vg[((size_t)b * ICc + m) * Tt + t0 + lr] = (__bf16)(accG[r] + bg[m]);
  }
}

// ---------------------------------------------------------------------------
// Flash attention: per wave a 16-query tile, block = 8 waves = 128 queries.
// Keys streamed in chunks of 32; K/V tiles double-buffered in LDS and filled
// by global_load_async_to_lds_b128 overlapping the previous chunk's WMMAs.
// ---------------------------------------------------------------------------
__global__ __launch_bounds__(256) void flash_kernel(
    const __bf16* __restrict__ Q, const __bf16* __restrict__ Kb,
    const __bf16* __restrict__ Vg, __bf16* __restrict__ O) {
  __shared__ __bf16 ldsK[2][32][ICc];   // 2 x 16 KB : keys   [key][channel]
  __shared__ __bf16 ldsV[2][ICc][32];   // 2 x 16 KB : values [channel][key]
  __shared__ __bf16 ldsP[8][16][32];    //     8 KB  : per-wave P staging

  const int tid  = threadIdx.x;
  const int lane = tid & 31, wave = tid >> 5;
  const int lr = lane & 15, h = lane >> 4;
  const int b  = blockIdx.x >> 4;                        // 16 blocks / batch
  const int q0 = (blockIdx.x & 15) * 128 + wave * 16;

  const __bf16* gK = Kb + (size_t)b * Tt * ICc;          // (T, IC)
  const __bf16* gV = Vg + (size_t)b * ICc * Tt;          // (IC, T)

  // async fill of one K/V chunk (32 keys) into buffer `buf`
  auto fill_chunk = [&](int buf, int j0) {
    const unsigned kb = lds_off(&ldsK[buf][0][0]);
    const unsigned vb = lds_off(&ldsV[buf][0][0]);
    const __bf16* kSrc = gK + (size_t)j0 * ICc;          // contiguous 16 KB
#pragma unroll
    for (int it = 0; it < 4; ++it) {
      const int i = tid + it * 256;                      // 1024 x 16B chunks
      async_ld_b128(kb + i * 16, (const char*)kSrc + i * 16);
    }
#pragma unroll
    for (int it = 0; it < 4; ++it) {
      const int i = tid + it * 256;
      const int row = i >> 2, col = i & 3;               // V row = channel
      async_ld_b128(vb + i * 16, gV + (size_t)row * Tt + j0 + col * 8);
    }
  };

  // Query fragments: 16 x 256 bf16 held in registers (A layout)
  v16bf qf[8];
  const __bf16* qrow = Q + ((size_t)b * Tt + q0 + lr) * ICc + 16 * h;
#pragma unroll
  for (int kc = 0; kc < 8; ++kc) qf[kc] = *(const v16bf*)(qrow + kc * 32);

  v8f acc[16] = {};
  float mrow[8], lrow[8];
#pragma unroll
  for (int r = 0; r < 8; ++r) { mrow[r] = -INFINITY; lrow[r] = 0.0f; }

  fill_chunk(0, 0);  // prefetch first chunk

  for (int c = 0; c < Tt / 32; ++c) {
    const int buf = c & 1;
    wait_async0();        // this wave's DMA for `buf` has landed
    __syncthreads();      // all waves' DMA landed; all done reading buf^1
    if (c + 1 < Tt / 32) fill_chunk(buf ^ 1, (c + 1) * 32);  // overlap w/ WMMAs

    // S = Q * K^T for two 16x16 key tiles (K dim = IC = 256)
    v8f s0 = {}, s1 = {};
#pragma unroll
    for (int kc = 0; kc < 8; ++kc) {
      v16bf b0 = *(const v16bf*)&ldsK[buf][lr][kc * 32 + 16 * h];
      v16bf b1 = *(const v16bf*)&ldsK[buf][16 + lr][kc * 32 + 16 * h];
      s0 = WMMA_BF16(qf[kc], b0, s0);
      s1 = WMMA_BF16(qf[kc], b1, s1);
    }

    // Online softmax update (rows live in 16-lane half-wave groups)
    float p0[8], p1[8];
#pragma unroll
    for (int r = 0; r < 8; ++r) {
      float mx = fmaxf(s0[r], s1[r]);
#pragma unroll
      for (int mk = 1; mk < 16; mk <<= 1) mx = fmaxf(mx, __shfl_xor(mx, mk, 16));
      const float mnew  = fmaxf(mrow[r], mx);
      const float scale = __expf(mrow[r] - mnew);
      const float e0 = __expf(s0[r] - mnew);
      const float e1 = __expf(s1[r] - mnew);
      float rs = e0 + e1;
#pragma unroll
      for (int mk = 1; mk < 16; mk <<= 1) rs += __shfl_xor(rs, mk, 16);
      lrow[r] = lrow[r] * scale + rs;
      mrow[r] = mnew;
      p0[r] = e0; p1[r] = e1;
#pragma unroll
      for (int nt = 0; nt < 16; ++nt) acc[nt][r] *= scale;
    }

    // P: D layout -> A layout via LDS round trip (wave-private region)
#pragma unroll
    for (int r = 0; r < 8; ++r) {
      ldsP[wave][r + 8 * h][lr]      = (__bf16)p0[r];
      ldsP[wave][r + 8 * h][16 + lr] = (__bf16)p1[r];
    }
    __syncthreads();
    v16bf pf = *(const v16bf*)&ldsP[wave][lr][16 * h];

    // O += P (16x32) * V (32x256): 16 N-tiles
#pragma unroll
    for (int nt = 0; nt < 16; ++nt) {
      v16bf bv = *(const v16bf*)&ldsV[buf][nt * 16 + lr][16 * h];
      acc[nt] = WMMA_BF16(pf, bv, acc[nt]);
    }
  }

#pragma unroll
  for (int r = 0; r < 8; ++r) lrow[r] = 1.0f / lrow[r];
#pragma unroll
  for (int nt = 0; nt < 16; ++nt)
#pragma unroll
    for (int r = 0; r < 8; ++r)
      O[((size_t)b * Tt + q0 + r + 8 * h) * ICc + nt * 16 + lr] =
          (__bf16)(acc[nt][r] * lrow[r]);
}

// ---------------------------------------------------------------------------
// wy = W_w (C x IC) * y (IC x T) + W_b, fp32 out (B,C,T)
// ---------------------------------------------------------------------------
__global__ __launch_bounds__(256) void wconv_kernel(
    const __bf16* __restrict__ Ww, const float* __restrict__ bias,
    const __bf16* __restrict__ yT, float* __restrict__ wy) {
  const int lane = threadIdx.x & 31;
  const int wave = threadIdx.x >> 5;
  const int wid  = blockIdx.x * 8 + wave;
  const int b    = wid >> 12;                // (C/16)*(T/16) = 4096 / batch
  const int rem  = wid & 4095;
  const int tIdx = rem >> 5;
  const int mIdx = rem & 31;
  const int m0 = mIdx * 16, t0 = tIdx * 16;
  const int lr = lane & 15, h = lane >> 4;

  v8f acc = {};
  const __bf16* arow = Ww + (size_t)(m0 + lr) * ICc + 16 * h;
  const __bf16* brow = yT + ((size_t)b * Tt + t0 + lr) * ICc + 16 * h;
#pragma unroll
  for (int k0 = 0; k0 < ICc; k0 += 32) {
    v16bf a  = *(const v16bf*)(arow + k0);
    v16bf bb = *(const v16bf*)(brow + k0);
    acc = WMMA_BF16(a, bb, acc);
  }
#pragma unroll
  for (int r = 0; r < 8; ++r) {
    const int m = m0 + 8 * h + r;
    wy[((size_t)b * Cc + m) * Tt + t0 + lr] = acc[r] + bias[m];
  }
}

// ---------------------------------------------------------------------------
// BatchNorm batch statistics: one block per channel
// ---------------------------------------------------------------------------
__global__ __launch_bounds__(256) void bnstats_kernel(
    const float* __restrict__ wy, float* __restrict__ mean,
    float* __restrict__ rstd) {
  const int c = blockIdx.x;
  float s = 0.0f, ss = 0.0f;
  for (int i = threadIdx.x; i < Bb * Tt; i += 256) {
    const int b = i >> 11, t = i & (Tt - 1);
    const float v = wy[((size_t)b * Cc + c) * Tt + t];
    s += v; ss += v * v;
  }
#pragma unroll
  for (int mk = 1; mk < 32; mk <<= 1) {
    s  += __shfl_xor(s, mk, 32);
    ss += __shfl_xor(ss, mk, 32);
  }
  __shared__ float red[2][8];
  const int wave = threadIdx.x >> 5;
  if ((threadIdx.x & 31) == 0) { red[0][wave] = s; red[1][wave] = ss; }
  __syncthreads();
  if (threadIdx.x == 0) {
    float S = 0.0f, SS = 0.0f;
#pragma unroll
    for (int w = 0; w < 8; ++w) { S += red[0][w]; SS += red[1][w]; }
    const float inv_n = 1.0f / (float)(Bb * Tt);
    const float mu  = S * inv_n;
    const float var = SS * inv_n - mu * mu;
    mean[c] = mu;
    rstd[c] = rsqrtf(var + 1e-5f);
  }
}

// ---------------------------------------------------------------------------
// out = relu((wy - mean) * rstd * gamma + beta) + x   (float4 vectorized)
// ---------------------------------------------------------------------------
__global__ __launch_bounds__(256) void finalize_kernel(
    const float* __restrict__ wy, const float* __restrict__ mean,
    const float* __restrict__ rstd, const float* __restrict__ gamma,
    const float* __restrict__ beta, const float* __restrict__ x,
    float* __restrict__ out) {
  const size_t i = ((size_t)blockIdx.x * blockDim.x + threadIdx.x) * 4;
  const int c = (int)((i / Tt) % Cc);
  const float4 w  = *(const float4*)(wy + i);
  const float4 xv = *(const float4*)(x + i);
  const float mu = mean[c], rs = rstd[c], g = gamma[c], bt = beta[c];
  float4 o;
  o.x = fmaxf(0.0f, (w.x - mu) * rs * g + bt) + xv.x;
  o.y = fmaxf(0.0f, (w.y - mu) * rs * g + bt) + xv.y;
  o.z = fmaxf(0.0f, (w.z - mu) * rs * g + bt) + xv.z;
  o.w = fmaxf(0.0f, (w.w - mu) * rs * g + bt) + xv.w;
  *(float4*)(out + i) = o;
}

// ---------------------------------------------------------------------------
extern "C" void kernel_launch(void* const* d_in, const int* in_sizes, int n_in,
                              void* d_out, int out_size, void* d_ws, size_t ws_size,
                              hipStream_t stream) {
  (void)in_sizes; (void)n_in; (void)out_size; (void)ws_size;

  const float* x       = (const float*)d_in[0];
  const float* g_w     = (const float*)d_in[1];
  const float* g_b     = (const float*)d_in[2];
  const float* theta_w = (const float*)d_in[3];
  const float* theta_b = (const float*)d_in[4];
  const float* phi_w   = (const float*)d_in[5];
  const float* phi_b   = (const float*)d_in[6];
  const float* W_w     = (const float*)d_in[7];
  const float* W_b     = (const float*)d_in[8];
  const float* bn_g    = (const float*)d_in[9];
  const float* bn_b    = (const float*)d_in[10];

  size_t off = 0;
  auto carve = [&](size_t bytes) -> void* {
    void* r = (char*)d_ws + off;
    off += (bytes + 255) & ~(size_t)255;
    return r;
  };
  __bf16* xT   = (__bf16*)carve((size_t)Bb * Tt * Cc * 2);   // 16 MB
  __bf16* gw16 = (__bf16*)carve((size_t)ICc * Cc * 2);
  __bf16* tw16 = (__bf16*)carve((size_t)ICc * Cc * 2);
  __bf16* pw16 = (__bf16*)carve((size_t)ICc * Cc * 2);
  __bf16* Ww16 = (__bf16*)carve((size_t)Cc * ICc * 2);
  __bf16* Q    = (__bf16*)carve((size_t)Bb * Tt * ICc * 2);  // 8 MB
  __bf16* Kb   = (__bf16*)carve((size_t)Bb * Tt * ICc * 2);  // 8 MB
  __bf16* Vg   = (__bf16*)carve((size_t)Bb * ICc * Tt * 2);  // 8 MB
  __bf16* yT   = (__bf16*)carve((size_t)Bb * Tt * ICc * 2);  // 8 MB
  float*  wy   = (float*) carve((size_t)Bb * Cc * Tt * 4);   // 32 MB
  float*  mean = (float*) carve((size_t)Cc * 4);
  float*  rstd = (float*) carve((size_t)Cc * 4);

  const int nW = ICc * Cc;

  f32_to_bf16_kernel<<<nW / 256, 256, 0, stream>>>(g_w, gw16, nW);
  f32_to_bf16_kernel<<<nW / 256, 256, 0, stream>>>(theta_w, tw16, nW);
  f32_to_bf16_kernel<<<nW / 256, 256, 0, stream>>>(phi_w, pw16, nW);
  f32_to_bf16_kernel<<<nW / 256, 256, 0, stream>>>(W_w, Ww16, nW);

  transpose_x_kernel<<<dim3(Tt / 32, Cc / 32, Bb), dim3(32, 8), 0, stream>>>(x, xT);

  proj_kernel<<<2048, 256, 0, stream>>>(xT, gw16, g_b, tw16, theta_b, pw16, phi_b,
                                        Q, Kb, Vg);

  flash_kernel<<<128, 256, 0, stream>>>(Q, Kb, Vg, yT);

  wconv_kernel<<<4096, 256, 0, stream>>>(Ww16, W_b, yT, wy);

  bnstats_kernel<<<Cc, 256, 0, stream>>>(wy, mean, rstd);

  finalize_kernel<<<(Bb * Cc * Tt / 4) / 256, 256, 0, stream>>>(
      wy, mean, rstd, bn_g, bn_b, x, (float*)d_out);
}
